// StackedBPLSTM_62826781606198
// MI455X (gfx1250) — matrix operand order repaired and checked by reading, not compiled, not run on hardware
//
#include <hip/hip_runtime.h>

// ============================================================================
// Stacked bidirectional peephole LSTM for MI455X (gfx1250, wave32, WMMA).
//
//  * pre = x@wih + b  : full-GPU bf16 WMMA GEMM over all (t,b) rows,
//    stored bf16 ([T,B,4H] per direction) in workspace.
//  * recurrence       : persistent kernel, 16 WGs per direction (cluster-sized),
//    whh slice [512 x 128] bf16 resident in LDS. Per-step staging of the h
//    panel and pre_t slice is done with the Tensor Data Mover
//    (tensor_load_to_lds + s_wait_tensorcnt) when the builtin is available,
//    else global->LDS copies. Grid sync: monotonic atomic barrier +
//    s_cluster_barrier + s_sleep spin.
//  * All matrix math in v_wmma_f32_16x16x32_bf16 (f32 accumulate).
// ============================================================================

#define Tt 1000
#define Bb 32
#define Dd 80
#define Hh 512
#define G4 2048   // 4*H

typedef __attribute__((ext_vector_type(16))) __bf16 v16bf;
typedef __attribute__((ext_vector_type(8)))  float  v8f;

union BF16Frag {
    unsigned short u[16];
    uint4 q[2];
    v16bf v;
};

__device__ __forceinline__ float bf2f(unsigned short u) {
    unsigned int x = ((unsigned int)u) << 16;
    return __uint_as_float(x);
}
__device__ __forceinline__ unsigned short f2bf(float f) {
    unsigned int u = __float_as_uint(f);
    unsigned int r = u + 0x7fffu + ((u >> 16) & 1u);   // round-to-nearest-even
    return (unsigned short)(r >> 16);
}
__device__ __forceinline__ float sigmoidf(float x) {
    return 1.0f / (1.0f + __expf(-x));
}

// ---------------------------------------------------------------------------
// Tensor Data Mover helper (gfx1250). Guarded: falls back to plain copies.
// D# layout per CDNA5 ISA ch.8 (group0: count/lds/global/type, group1: dims).
// ---------------------------------------------------------------------------
#if __has_builtin(__builtin_amdgcn_tensor_load_to_lds)
#define LSTM_HAVE_TDM 1
typedef __attribute__((ext_vector_type(4))) unsigned int u32x4;
typedef __attribute__((ext_vector_type(4))) int i32x4;
typedef __attribute__((ext_vector_type(8))) int i32x8;

// Loads a 3D tile of 2-byte elements:
//   X: tile0 contiguous elems; Y: tile1 lines at stride0 elems;
//   Z: tile2 planes at stride1 elems.  LDS fill is linear from lds_byte_off.
__device__ __forceinline__ void tdm_load_tile_to_lds(
    unsigned lds_byte_off, const void* gptr,
    unsigned tile0, unsigned tile1, unsigned tile2,
    unsigned long long stride0_elems, unsigned long long stride1_elems)
{
    unsigned long long ga = (unsigned long long)(size_t)gptr;
    u32x4 g0;
    g0[0] = 1u;                                            // count=1 (user D#)
    g0[1] = lds_byte_off;                                  // lds_addr (bytes)
    g0[2] = (unsigned)ga;                                  // global_addr[31:0]
    g0[3] = (unsigned)((ga >> 32) & 0x01ffffffu) | (2u << 30);  // [56:32]|type=2
    const unsigned td0 = 0x7fffffffu, td1 = 0x7fffffffu;   // huge dims: no clip
    i32x8 g1;
    g1[0] = (int)(1u << 16);                               // data_size=1 (2B)
    g1[1] = (int)((td0 & 0xffffu) << 16);                  // tensor_dim0 lo
    g1[2] = (int)(((td0 >> 16) & 0xffffu) | ((td1 & 0xffffu) << 16));
    g1[3] = (int)(((td1 >> 16) & 0xffffu) | ((tile0 & 0xffffu) << 16));
    g1[4] = (int)((tile1 & 0xffffu) | ((tile2 & 0xffffu) << 16));
    g1[5] = (int)(unsigned)(stride0_elems & 0xffffffffull);
    g1[6] = (int)(((unsigned)(stride0_elems >> 32) & 0xffffu) |
                  (((unsigned)stride1_elems & 0xffffu) << 16));
    g1[7] = (int)(unsigned)((stride1_elems >> 16) & 0xffffffffull);
    i32x4 g2 = {(int)0x7fffffff, 0, 0, 0};  // tensor_dim2 huge; no dim3/iterate
    i32x4 g3 = {0, 0, 0, 0};
#if __clang_major__ >= 23
    i32x8 g4 = {0, 0, 0, 0, 0, 0, 0, 0};
    __builtin_amdgcn_tensor_load_to_lds(g0, g1, g2, g3, g4, 0);
#else
    __builtin_amdgcn_tensor_load_to_lds(g0, g1, g2, g3, 0);
#endif
}
#endif  // has tensor_load_to_lds

// ---------------------------------------------------------------------------
// f32 [B,T,80] -> bf16 [B,T,96] (zero-padded K so WMMA K-loop divides by 32)
// ---------------------------------------------------------------------------
__global__ void convert_x0_kernel(const float* __restrict__ x,
                                  unsigned short* __restrict__ xbf) {
    const size_t N = (size_t)Bb * Tt * 96;
    for (size_t e = (size_t)blockIdx.x * blockDim.x + threadIdx.x; e < N;
         e += (size_t)gridDim.x * blockDim.x) {
        size_t row = e / 96;
        int k = (int)(e % 96);
        xbf[e] = (k < Dd) ? f2bf(x[row * Dd + k]) : (unsigned short)0;
    }
}

// ---------------------------------------------------------------------------
// Weight convert + transpose: f32 [K, 2048] -> bf16 [2048, Kpad] (zero pad)
// ---------------------------------------------------------------------------
__global__ void convert_wT_kernel(const float* __restrict__ w,
                                  unsigned short* __restrict__ wT,
                                  int K, int Kpad) {
    const size_t N = (size_t)G4 * Kpad;
    for (size_t e = (size_t)blockIdx.x * blockDim.x + threadIdx.x; e < N;
         e += (size_t)gridDim.x * blockDim.x) {
        int n = (int)(e / Kpad);
        int k = (int)(e % Kpad);
        wT[e] = (k < K) ? f2bf(w[(size_t)k * G4 + n]) : (unsigned short)0;
    }
}

__global__ void zero_u32_kernel(unsigned int* __restrict__ p, size_t n) {
    for (size_t e = (size_t)blockIdx.x * blockDim.x + threadIdx.x; e < n;
         e += (size_t)gridDim.x * blockDim.x)
        p[e] = 0u;
}

// ---------------------------------------------------------------------------
// pre[dir][t,b,:] = x[b, tphys(dir,t), :] @ wih + bias     (bf16 WMMA, f32 acc)
// ---------------------------------------------------------------------------
__global__ __launch_bounds__(256) void pregemm_kernel(
    const unsigned short* __restrict__ xbf,   // [B,T,Kpad] bf16
    const unsigned short* __restrict__ wT_f,  // [2048,Kpad] bf16
    const unsigned short* __restrict__ wT_b,
    const float* __restrict__ bias_f,         // [2048] f32
    const float* __restrict__ bias_b,
    unsigned short* __restrict__ pre,         // [2][T*B*2048] bf16
    int Kpad) {
    const int tid = threadIdx.x;
    const int lane = tid & 31;
    const size_t wid = (size_t)blockIdx.x * 8 + (tid >> 5);
    const int MT = (Tt * Bb) / 16;  // 2000
    const int NT = G4 / 16;         // 128
    const size_t per_dir = (size_t)MT * NT;
    const int dir = (int)(wid / per_dir);
    const size_t rem = wid % per_dir;
    const int mt = (int)(rem / NT), nt = (int)(rem % NT);

    const unsigned short* wT = dir ? wT_b : wT_f;
    const float* bias = dir ? bias_b : bias_f;

    const int rlo = lane & 15;
    const int hi = (lane >= 16) ? 1 : 0;
    const int kb_a = hi ? 8 : 0;
    const int kb_b = hi ? 8 : 0;
    const int n = nt * 16 + rlo;

    const int R = mt * 16 + rlo;
    const int tlog = R >> 5, b = R & 31;
    const int tphys = dir ? (Tt - 1 - tlog) : tlog;
    const unsigned short* arow = xbf + ((size_t)b * Tt + tphys) * Kpad;
    const unsigned short* brow = wT + (size_t)n * Kpad;

    v8f acc = {0, 0, 0, 0, 0, 0, 0, 0};
    for (int k0 = 0; k0 < Kpad; k0 += 32) {
        BF16Frag fa, fb;
        fa.q[0] = *(const uint4*)(arow + k0 + kb_a);
        fa.q[1] = *(const uint4*)(arow + k0 + kb_a + 16);
        fb.q[0] = *(const uint4*)(brow + k0 + kb_b);
        fb.q[1] = *(const uint4*)(brow + k0 + kb_b + 16);
        acc = __builtin_amdgcn_wmma_f32_16x16x32_bf16(
            false, fa.v, false, fb.v, (short)0, acc, false, false);
    }

    const float bn = bias[n];
    unsigned short* pdir = pre + (size_t)dir * Tt * Bb * G4;
#pragma unroll
    for (int r = 0; r < 8; ++r) {
        int m = mt * 16 + r + (hi ? 8 : 0);
        int tl = m >> 5, bb = m & 31;
        pdir[((size_t)tl * Bb + bb) * G4 + n] = f2bf(acc[r] + bn);
    }
}

// ---------------------------------------------------------------------------
// Persistent bidirectional LSTM scan.
// Grid = 32 WGs x 256 threads. WG 0..15 forward, 16..31 backward.
// LDS layout (dynamic, base offset 0):
//   Bp   [128][512] bf16  @ 0        (whh slice, resident, 128 KB)
//   Ap   [ 32][512] bf16  @ 131072   (h panel, 32 KB)
//   Zpre [ 32][128] bf16  @ 163840   (pre_t slice, 8 KB)
//   Z    [ 32][128] f32   @ 172032   (gemm result, 16 KB)
//   Cst  [ 32][ 32] f32   @ 188416   (cell state, 4 KB)
// ---------------------------------------------------------------------------
#define BP_OFF   0u
#define AP_OFF   (128u * 512u * 2u)
#define ZPRE_OFF (AP_OFF + 32u * 512u * 2u)

__global__ __launch_bounds__(256) void lstm_scan_kernel(
    const unsigned short* __restrict__ pre,     // [2][T*B*2048] bf16
    const unsigned short* __restrict__ whhT_f,  // [2048][512] bf16 (transposed)
    const unsigned short* __restrict__ whhT_b,
    const float* __restrict__ wch_f,            // [3H] peephole
    const float* __restrict__ wch_b,
    const int* __restrict__ lengths,            // [B]
    unsigned short* __restrict__ hbuf,          // [2 dir][2 pingpong][B*H] bf16
    unsigned int* __restrict__ bar,             // [2 dir x 16 u32 spacing]
    unsigned short* __restrict__ out_bf,        // [B,T,2H] bf16 or null
    float* __restrict__ out_f32)                // [B,T,2H] f32 or null
{
    extern __shared__ char smem_raw[];
    unsigned short* Bp   = (unsigned short*)smem_raw;   // [128][512]
    unsigned short* Ap   = Bp + 128 * 512;              // [32][512]
    unsigned short* Zpre = Ap + 32 * 512;               // [32][128]
    float* Z   = (float*)(Zpre + 32 * 128);             // [32][128]
    float* Cst = Z + 32 * 128;                          // [32][32]

    const int tid = threadIdx.x;
    const int wg = blockIdx.x;
    const int dir = wg >> 4;
    const int g = wg & 15;
    const unsigned short* whhT = dir ? whhT_b : whhT_f;
    const float* wch = dir ? wch_b : wch_f;
    const unsigned short* pdir = pre + (size_t)dir * Tt * Bb * G4;
    unsigned int* mybar = bar + dir * 16;

    // ---- resident whh slice: 4 gate blocks of 32 contiguous rows ----------
#ifdef LSTM_HAVE_TDM
    if (tid < 32) {
        tdm_load_tile_to_lds(BP_OFF, whhT + (size_t)(g * 32) * 512,
                             /*X*/ 16384u, /*Y gates*/ 4u, /*Z*/ 0u,
                             /*strideY*/ 512ull * 512ull, 0ull);
        __builtin_amdgcn_s_wait_tensorcnt(0);
    }
#else
    for (int c = tid; c < 128 * 64; c += 256) {
        int ln = c >> 6, ch = c & 63;
        int n = (ln >> 5) * Hh + g * 32 + (ln & 31);
        ((uint4*)(Bp + ln * 512))[ch] = ((const uint4*)(whhT + (size_t)n * 512))[ch];
    }
#endif
    for (int e = tid; e < 1024; e += 256) Cst[e] = 0.0f;

    const int lane = tid & 31;
    const int wave = tid >> 5;
    const int hi = (lane >= 16) ? 1 : 0;
    const int rlo = lane & 15;
    const int kb_a = hi ? 8 : 0;
    const int kb_b = hi ? 8 : 0;
    const int nloc = wave * 16 + rlo;  // local gate column 0..127

    for (int t = 0; t < Tt; ++t) {
        // ---- stage h_{t} panel and pre_t slice into LDS -------------------
        const unsigned short* hsrc = hbuf + ((size_t)dir * 2 + (t & 1)) * Bb * Hh;
#ifdef LSTM_HAVE_TDM
        if (tid < 32) {
            // h panel: one contiguous 32 KB tile
            tdm_load_tile_to_lds(AP_OFF, hsrc, 16384u, 1u, 0u, 0ull, 0ull);
            // pre slice: X=32 elems, Y=4 gates (stride 512), Z=32 batch (2048)
            tdm_load_tile_to_lds(ZPRE_OFF, pdir + (size_t)t * Bb * G4 + g * 32,
                                 32u, 4u, 32u, 512ull, 2048ull);
            __builtin_amdgcn_s_wait_tensorcnt(0);
        }
#else
        for (int c = tid; c < 2048; c += 256)
            ((uint4*)Ap)[c] = ((const uint4*)hsrc)[c];
        for (int c = tid; c < 512; c += 256) {
            int b = c >> 4, rm = c & 15, q = rm >> 2, p = rm & 3;
            *(uint4*)(Zpre + b * 128 + q * 32 + p * 8) =
                *(const uint4*)(pdir + ((size_t)t * Bb + b) * G4 + q * Hh + g * 32 + p * 8);
        }
#endif
        __syncthreads();

        // ---- z = h @ whh_slice : [32,512] x [512,128] in bf16 WMMA --------
        v8f acc0 = {0, 0, 0, 0, 0, 0, 0, 0};
        v8f acc1 = {0, 0, 0, 0, 0, 0, 0, 0};
        const unsigned short* brow  = Bp + nloc * 512;
        const unsigned short* arow0 = Ap + rlo * 512;
        const unsigned short* arow1 = Ap + (16 + rlo) * 512;
        for (int k0 = 0; k0 < Hh; k0 += 32) {
            BF16Frag fb, fa0, fa1;
            fb.q[0]  = *(const uint4*)(brow + k0 + kb_b);
            fb.q[1]  = *(const uint4*)(brow + k0 + kb_b + 16);
            fa0.q[0] = *(const uint4*)(arow0 + k0 + kb_a);
            fa0.q[1] = *(const uint4*)(arow0 + k0 + kb_a + 16);
            fa1.q[0] = *(const uint4*)(arow1 + k0 + kb_a);
            fa1.q[1] = *(const uint4*)(arow1 + k0 + kb_a + 16);
            acc0 = __builtin_amdgcn_wmma_f32_16x16x32_bf16(
                false, fa0.v, false, fb.v, (short)0, acc0, false, false);
            acc1 = __builtin_amdgcn_wmma_f32_16x16x32_bf16(
                false, fa1.v, false, fb.v, (short)0, acc1, false, false);
        }
#pragma unroll
        for (int r = 0; r < 8; ++r) {
            int m = r + (hi ? 8 : 0);
            Z[m * 128 + nloc]        = acc0[r];
            Z[(16 + m) * 128 + nloc] = acc1[r];
        }
        __syncthreads();

        // ---- gates / cell / hidden update ---------------------------------
        unsigned short* hdst = hbuf + ((size_t)dir * 2 + ((t + 1) & 1)) * Bb * Hh;
#pragma unroll
        for (int r = 0; r < 4; ++r) {
            int e = tid + r * 256;       // 0..1023
            int b = e >> 5, j = e & 31;
            int hj = g * 32 + j;
            float zf = Z[b * 128 + j]       + bf2f(Zpre[b * 128 + j]);
            float zi = Z[b * 128 + 32 + j]  + bf2f(Zpre[b * 128 + 32 + j]);
            float zg = Z[b * 128 + 64 + j]  + bf2f(Zpre[b * 128 + 64 + j]);
            float zo = Z[b * 128 + 96 + j]  + bf2f(Zpre[b * 128 + 96 + j]);
            float c_old = Cst[e];
            float fg = sigmoidf(zf + c_old * wch[hj]);
            float ig = sigmoidf(zi + c_old * wch[Hh + hj]);
            float gg = tanhf(zg);
            float cn = fg * c_old + ig * gg;
            float og = sigmoidf(zo + cn * wch[2 * Hh + hj]);
            float m = (dir == 0) ? 1.0f
                                 : ((lengths[b] >= (Tt - t - 1)) ? 1.0f : 0.0f);
            float hn = og * tanhf(cn) * m;
            Cst[e] = cn;
            hdst[b * Hh + hj] = f2bf(hn);
            int tphys = dir ? (Tt - 1 - t) : t;
            size_t oidx = ((size_t)b * Tt + tphys) * (2 * Hh) + dir * Hh + hj;
            if (out_f32) out_f32[oidx] = hn;
            else         out_bf[oidx]  = f2bf(hn);
        }

        // Prefetch next step's pre slice into cache (global_prefetch_b8).
        if (t + 1 < Tt)
            __builtin_prefetch(
                pdir + ((size_t)(t + 1) * Bb + (tid >> 3)) * G4 + g * 32 + (tid & 7) * 4,
                0, 1);

        // ---- direction-wide barrier (16 WGs) ------------------------------
        __syncthreads();
        if (tid == 0) {
            __threadfence();
            unsigned arrive = __hip_atomic_fetch_add(
                mybar, 1u, __ATOMIC_ACQ_REL, __HIP_MEMORY_SCOPE_AGENT);
            unsigned target = (arrive / 16u + 1u) * 16u;
            while (__hip_atomic_load(mybar, __ATOMIC_ACQUIRE,
                                     __HIP_MEMORY_SCOPE_AGENT) < target)
                __builtin_amdgcn_s_sleep(1);
        }
        // Fast path if dispatched as a 16-WG cluster; NOP otherwise.
        if (tid < 32) __builtin_amdgcn_s_cluster_barrier();
        __syncthreads();
    }
}

// ===========================================================================
// Host launcher
// ===========================================================================
extern "C" void kernel_launch(void* const* d_in, const int* in_sizes, int n_in,
                              void* d_out, int out_size, void* d_ws, size_t ws_size,
                              hipStream_t stream) {
    (void)in_sizes; (void)n_in; (void)out_size; (void)ws_size;
    const float* x0 = (const float*)d_in[0];
    const int* lengths = (const int*)d_in[1];
    // Per layer param order: wih_f, whh_f, wch_f, b_f, wih_b, whh_b, wch_b, b_b
    auto P = [&](int l, int idx) -> const float* {
        return (const float*)d_in[2 + l * 8 + idx];
    };

    char* ws = (char*)d_ws;
    size_t off = 0;
    auto alloc = [&](size_t bytes) -> char* {
        char* p = ws + off;
        off = (off + bytes + 255) & ~(size_t)255;
        return p;
    };

    const int Kpad[3] = {96, 1024, 1024};
    unsigned short* wihT[3][2];
    unsigned short* whhT[3][2];
    for (int l = 0; l < 3; ++l)
        for (int d = 0; d < 2; ++d) {
            wihT[l][d] = (unsigned short*)alloc((size_t)G4 * Kpad[l] * 2);
            whhT[l][d] = (unsigned short*)alloc((size_t)G4 * Hh * 2);
        }
    unsigned short* xa  = (unsigned short*)alloc((size_t)Bb * Tt * 1024 * 2);
    unsigned short* xb  = (unsigned short*)alloc((size_t)Bb * Tt * 1024 * 2);
    unsigned short* pre = (unsigned short*)alloc((size_t)2 * Tt * Bb * G4 * 2);
    unsigned short* hbuf = (unsigned short*)alloc((size_t)2 * 2 * Bb * Hh * 2);
    unsigned int* bar = (unsigned int*)alloc(128);

    convert_x0_kernel<<<256, 256, 0, stream>>>(x0, xa);
    for (int l = 0; l < 3; ++l) {
        int K = (l == 0) ? Dd : 1024;
        convert_wT_kernel<<<512, 256, 0, stream>>>(P(l, 0), wihT[l][0], K, Kpad[l]);
        convert_wT_kernel<<<512, 256, 0, stream>>>(P(l, 4), wihT[l][1], K, Kpad[l]);
        convert_wT_kernel<<<512, 256, 0, stream>>>(P(l, 1), whhT[l][0], Hh, Hh);
        convert_wT_kernel<<<512, 256, 0, stream>>>(P(l, 5), whhT[l][1], Hh, Hh);
    }

    const size_t smem = (size_t)128 * 512 * 2 + 32 * 512 * 2 + 32 * 128 * 2 +
                        32 * 128 * 4 + 32 * 32 * 4;  // ~188 KB
    hipFuncSetAttribute(reinterpret_cast<const void*>(lstm_scan_kernel),
                        hipFuncAttributeMaxDynamicSharedMemorySize, (int)smem);

    unsigned short* cur = xa;
    unsigned short* nxt = xb;
    for (int l = 0; l < 3; ++l) {
        zero_u32_kernel<<<64, 256, 0, stream>>>((unsigned int*)hbuf,
                                                (size_t)2 * 2 * Bb * Hh * 2 / 4);
        zero_u32_kernel<<<1, 32, 0, stream>>>(bar, 32);

        const size_t nwaves = (size_t)2 * ((size_t)Tt * Bb / 16) * (G4 / 16);
        pregemm_kernel<<<(int)(nwaves / 8), 256, 0, stream>>>(
            cur, wihT[l][0], wihT[l][1], P(l, 3), P(l, 7), pre, Kpad[l]);

        const bool final_layer = (l == 2);
        lstm_scan_kernel<<<32, 256, smem, stream>>>(
            pre, whhT[l][0], whhT[l][1], P(l, 2), P(l, 6), lengths, hbuf, bar,
            final_layer ? nullptr : nxt,
            final_layer ? (float*)d_out : nullptr);

        unsigned short* tmp = cur; cur = nxt; nxt = tmp;
    }
}
